// FunctionEncoder_72344429134414
// MI455X (gfx1250) — compile-verified, matrix operand order checked by reading, not run on hardware
//
#include <hip/hip_runtime.h>
#include <hip/hip_bf16.h>

typedef _Float16 half_t;
typedef __attribute__((ext_vector_type(16))) _Float16 v16h;
typedef __attribute__((ext_vector_type(8)))  float    v8f;

#define BS   65536
#define LL   32
#define NC   10
#define KK   4
#define TT   8
#define EMB  256
#define ZD   128
#define NE   128
#define PP   80    // NC*TT
#define WPB  4     // waves per block

// ---------------- init: zero global accumulators ----------------
__global__ void init_kernel(float* g_cnt, float* g_sq, float* g_vd) {
    int t = threadIdx.x;
    if (t < NE) g_cnt[t] = 0.f;
    if (t == 128) *g_sq = 0.f;
    if (t == 129) *g_vd = 0.f;
}

// ---------------- precompute: W2=mu_w@fc_w, b2, cbn, swizzled B operands ----
__global__ __launch_bounds__(256) void precomp_kernel(
    const float* __restrict__ codebook, const float* __restrict__ fc_w,
    const float* __restrict__ fc_b, const float* __restrict__ mu_w,
    const float* __restrict__ mu_b,
    half_t* __restrict__ w2B, half_t* __restrict__ cbB,
    float* __restrict__ b2, float* __restrict__ cbn)
{
    __shared__ float sW2[ZD * PP];            // 40 KB
    const int tid = threadIdx.x;

    for (int i = tid; i < ZD * PP; i += 256) {
        int z = i / PP, p = i % PP;
        float acc = 0.f;
        for (int e = 0; e < EMB; ++e)
            acc = fmaf(mu_w[z * EMB + e], fc_w[e * PP + p], acc);
        sW2[i] = acc;
    }
    if (tid < ZD) {
        float acc = mu_b[tid];
        for (int e = 0; e < EMB; ++e)
            acc = fmaf(mu_w[tid * EMB + e], fc_b[e], acc);
        b2[tid] = acc;
    } else {
        int e = tid - 128;                    // NE == 128
        float acc = 0.f;
        for (int c = 0; c < NC; ++c) { float v = codebook[e * NC + c]; acc = fmaf(v, v, acc); }
        cbn[e] = acc;
    }
    __syncthreads();

    // w2B[(s*8+nt)*32+lane][j] : B operand (32x16, K x N), K=p padded to 96, N=z
    for (int i = tid; i < 3 * 8 * 512; i += 256) {
        int op = i >> 9, li = i & 511;
        int lane = li >> 4, j = li & 15;
        int s = op >> 3, nt = op & 7;
        int n = nt * 16 + (lane & 15);
        int k = (lane < 16) ? (s * 32 + ((j < 8) ? j : 16 + (j - 8)))
                            : (s * 32 + ((j < 8) ? 8 + j : 24 + (j - 8)));
        w2B[i] = (half_t)((k < PP) ? sW2[n * PP + k] : 0.f);
    }
    // cbB[et*32+lane][j] : B operand for VQ, K=channel (10 pad 32), N=entry, value -2*cb
    for (int i = tid; i < 8 * 512; i += 256) {
        int et = i >> 9, li = i & 511;
        int lane = li >> 4, j = li & 15;
        int n = et * 16 + (lane & 15);
        int k = (lane < 16) ? ((j < 8) ? j : 16 + (j - 8))
                            : ((j < 8) ? 8 + j : 24 + (j - 8));
        cbB[i] = (half_t)((k < NC) ? (-2.f * codebook[n * NC + k]) : 0.f);
    }
}

// ---------------- main fused kernel ----------------
__global__ __launch_bounds__(128) void vqenc_kernel(
    const float* __restrict__ fn, const unsigned char* __restrict__ pmask,
    const float* __restrict__ conv_w, const float* __restrict__ conv_b,
    const float* __restrict__ codebook,
    const half_t* __restrict__ w2B, const half_t* __restrict__ cbB,
    const float* __restrict__ b2g, const float* __restrict__ cbng,
    float* __restrict__ g_cnt, float* __restrict__ g_sq, float* __restrict__ g_vd,
    float* __restrict__ out)
{
    __shared__ float  s_cb[NE * NC];          // codebook f32 (gathers)
    __shared__ float  s_cw[NC * KK];
    __shared__ float  s_cbv[NC];
    __shared__ float  s_b2[ZD];
    __shared__ float  s_cbn[NE];
    __shared__ half_t s_zh[WPB][16 * PP];     // z (f16), per wave
    __shared__ float  s_sc[WPB][16 * NE];     // VQ scores, per wave (reused per A-tile)
    __shared__ int    s_idx[WPB][16 * TT];    // codebook indices, per wave
    __shared__ float  s_cnt[NE];
    __shared__ float  s_red[2];

    const int tid  = threadIdx.x;
    const int w    = tid >> 5;
    const int lane = tid & 31;
    const int m    = lane & 15;

    for (int i = tid; i < NE * NC; i += 128) s_cb[i] = codebook[i];
    if (tid < NC * KK) s_cw[tid] = conv_w[tid];
    if (tid < NC)      s_cbv[tid] = conv_b[tid];
    if (tid < ZD)      s_b2[tid] = b2g[tid];
    if (tid < NE) { s_cbn[tid] = cbng[tid]; s_cnt[tid] = 0.f; }
    if (tid < 2)  s_red[tid] = 0.f;
    __syncthreads();

    const int rowbase = (blockIdx.x * WPB + w) * 16;

    // ---- conv1d(k=4,s=4) + ReLU -> z in LDS (f16) ----
    #pragma unroll
    for (int j = 0; j < 4; ++j) {
        int rt = lane + 32 * j;
        int row = rt >> 3, t = rt & 7;
        const float4 v = *(const float4*)(fn + (size_t)(rowbase + row) * LL + t * KK);
        #pragma unroll
        for (int c = 0; c < NC; ++c) {
            float a = s_cbv[c];
            a = fmaf(v.x, s_cw[c * 4 + 0], a);
            a = fmaf(v.y, s_cw[c * 4 + 1], a);
            a = fmaf(v.z, s_cw[c * 4 + 2], a);
            a = fmaf(v.w, s_cw[c * 4 + 3], a);
            s_zh[w][row * PP + t * NC + c] = (half_t)fmaxf(a, 0.f);
        }
    }
    __builtin_amdgcn_wave_barrier();

    // ---- preload VQ B operands (one 32B v16h load per lane per operand) ----
    v16h cbBreg[8];
    #pragma unroll
    for (int et = 0; et < 8; ++et)
        cbBreg[et] = *(const v16h*)(cbB + (et * 32 + lane) * 16);

    // ---- VQ: scores via WMMA, argmin per (row,t) ----
    for (int a = 0; a < 8; ++a) {
        int row2 = 2 * a + (m >> 3);
        int tt   = m & 7;
        const half_t* zp = &s_zh[w][row2 * PP + tt * NC];
        v16h A;
        #pragma unroll
        for (int j = 0; j < 8; ++j) {
            half_t lo = zp[j];
            half_t hi = (j < 2) ? zp[8 + j] : (half_t)0.f;
            A[j] = (lane < 16) ? lo : hi;
        }
        #pragma unroll
        for (int j = 8; j < 16; ++j) A[j] = (half_t)0.f;

        const int rbase = (lane < 16) ? 0 : 8;
        #pragma unroll
        for (int et = 0; et < 8; ++et) {
            float ci = s_cbn[et * 16 + m];
            v8f C;
            #pragma unroll
            for (int r = 0; r < 8; ++r) C[r] = ci;
            v8f D = __builtin_amdgcn_wmma_f32_16x16x32_f16(
                false, A, false, cbBreg[et], (short)0, C, false, false);
            #pragma unroll
            for (int r = 0; r < 8; ++r)
                s_sc[w][(rbase + r) * NE + et * 16 + m] = D[r];
        }
        __builtin_amdgcn_wave_barrier();

        // two lanes per row: lane m scans e 0..63, lane m+16 scans 64..127
        int ebase = (lane < 16) ? 0 : 64;
        float bv = 3.0e38f; int bi = ebase;
        const float* sp = &s_sc[w][m * NE + ebase];
        for (int e = 0; e < 64; ++e) {
            float v = sp[e];
            if (v < bv) { bv = v; bi = ebase + e; }
        }
        float ov = __shfl_xor(bv, 16, 32);
        int   oi = __shfl_xor(bi, 16, 32);
        if (lane < 16) {
            if (ov < bv) { bv = ov; bi = oi; }          // tie -> keep lower-index half
            s_idx[w][(2 * a + (m >> 3)) * TT + (m & 7)] = bi;
        }
        __builtin_amdgcn_wave_barrier();
    }

    // ---- commitment loss + codebook counts ----
    float lsq = 0.f, lvd = 0.f;
    #pragma unroll
    for (int j = 0; j < 4; ++j) {
        int rt = lane + 32 * j;
        int row = rt >> 3, t = rt & 7;
        int e = s_idx[w][row * TT + t];
        float vb = pmask[rowbase + row] ? 0.f : 1.f;
        const float*  cbp = &s_cb[e * NC];
        const half_t* zp  = &s_zh[w][row * PP + t * NC];
        float err = 0.f;
        #pragma unroll
        for (int c = 0; c < NC; ++c) {
            float d = cbp[c] - (float)zp[c];
            err = fmaf(d, d, err);
        }
        lsq += err * vb;
        lvd += vb;
        atomicAdd(&s_cnt[e], vb);
    }
    atomicAdd(&s_red[0], lsq);
    atomicAdd(&s_red[1], lvd);

    // ---- output GEMM: out = zf(q) @ W2^T + b2, K=80 pad 96 ----
    v16h Ao[3];
    #pragma unroll
    for (int s = 0; s < 3; ++s) {
        #pragma unroll
        for (int j = 0; j < 16; ++j) {
            const int klo = s * 32 + ((j < 8) ? j : 16 + (j - 8));
            const int khi = s * 32 + ((j < 8) ? 8 + j : 24 + (j - 8));
            half_t vlo = (half_t)0.f, vhi = (half_t)0.f;
            if (klo < PP) { int c = klo >> 3, t = klo & 7; vlo = (half_t)s_cb[s_idx[w][m * TT + t] * NC + c]; }
            if (khi < PP) { int c = khi >> 3, t = khi & 7; vhi = (half_t)s_cb[s_idx[w][m * TT + t] * NC + c]; }
            Ao[s][j] = (lane < 16) ? vlo : vhi;
        }
    }
    float* outp = out + (size_t)rowbase * ZD;
    const int rbase = (lane < 16) ? 0 : 8;
    #pragma unroll
    for (int nt = 0; nt < 8; ++nt) {
        float bi = s_b2[nt * 16 + m];
        v8f acc;
        #pragma unroll
        for (int r = 0; r < 8; ++r) acc[r] = bi;
        #pragma unroll
        for (int s = 0; s < 3; ++s) {
            v16h Bv = *(const v16h*)(w2B + ((s * 8 + nt) * 32 + lane) * 16);
            acc = __builtin_amdgcn_wmma_f32_16x16x32_f16(
                false, Ao[s], false, Bv, (short)0, acc, false, false);
        }
        #pragma unroll
        for (int r = 0; r < 8; ++r)
            outp[(size_t)(rbase + r) * ZD + nt * 16 + m] = acc[r];
    }

    // ---- block-level flush of reductions ----
    __syncthreads();
    if (tid < NE) {
        float c = s_cnt[tid];
        if (c != 0.f) atomicAdd(&g_cnt[tid], c);
    }
    if (tid == 0) atomicAdd(g_sq, s_red[0]);
    if (tid == 1) atomicAdd(g_vd, s_red[1]);
}

// ---------------- finalize scalars ----------------
__global__ __launch_bounds__(128) void finalize_kernel(
    const float* g_cnt, const float* g_sq, const float* g_vd, float* out_tail)
{
    __shared__ float sv[NE];
    int t = threadIdx.x;
    sv[t] = g_cnt[t];
    __syncthreads();
    if (t == 0) {
        float tot = 0.f;
        for (int i = 0; i < NE; ++i) tot += sv[i];
        float ent = 0.f;
        for (int i = 0; i < NE; ++i) {
            float p = sv[i] / tot;
            ent += p * logf(p + 1e-10f);
        }
        out_tail[0] = 0.25f * (*g_sq) / ((*g_vd) * (float)NC);  // commitment loss
        out_tail[1] = expf(-ent);                                // perplexity
    }
}

extern "C" void kernel_launch(void* const* d_in, const int* in_sizes, int n_in,
                              void* d_out, int out_size, void* d_ws, size_t ws_size,
                              hipStream_t stream)
{
    const float*         fn       = (const float*)d_in[0];
    const unsigned char* pmask    = (const unsigned char*)d_in[1];
    const float*         conv_w   = (const float*)d_in[2];
    const float*         conv_b   = (const float*)d_in[3];
    const float*         codebook = (const float*)d_in[4];
    const float*         fc_w     = (const float*)d_in[5];
    const float*         fc_b     = (const float*)d_in[6];
    const float*         mu_w     = (const float*)d_in[7];
    const float*         mu_b     = (const float*)d_in[8];
    float* out = (float*)d_out;

    char* ws = (char*)d_ws;
    half_t* w2B   = (half_t*)(ws);            // 24576 B
    half_t* cbB   = (half_t*)(ws + 24576);    //  8192 B
    float*  b2    = (float*) (ws + 32768);    //   512 B
    float*  cbn   = (float*) (ws + 33280);    //   512 B
    float*  g_cnt = (float*) (ws + 33792);    //   512 B
    float*  g_sq  = (float*) (ws + 34304);
    float*  g_vd  = (float*) (ws + 34308);

    init_kernel<<<1, 256, 0, stream>>>(g_cnt, g_sq, g_vd);
    precomp_kernel<<<1, 256, 0, stream>>>(codebook, fc_w, fc_b, mu_w, mu_b,
                                          w2B, cbB, b2, cbn);
    vqenc_kernel<<<BS / (WPB * 16), 128, 0, stream>>>(
        fn, pmask, conv_w, conv_b, codebook, w2B, cbB, b2, cbn,
        g_cnt, g_sq, g_vd, out);
    finalize_kernel<<<1, 128, 0, stream>>>(g_cnt, g_sq, g_vd, out + (size_t)BS * ZD);
}
